// sarResNet_4715874091681
// MI455X (gfx1250) — matrix operand order repaired and checked by reading, not compiled, not run on hardware
//
#include <hip/hip_runtime.h>
#include <hip/hip_bf16.h>
#include <math.h>

// ---------------------------------------------------------------------------
// CDNA5 (gfx1250) WMMA types
// ---------------------------------------------------------------------------
typedef __attribute__((ext_vector_type(16))) __bf16 v16bf;
typedef __attribute__((ext_vector_type(8)))  float  v8f;

#define EPI_RELU  1
#define EPI_RESID 2
#define EPI_MASK  4

__device__ inline v8f wmma_bf16(v16bf a, v16bf b, v8f c) {
  return __builtin_amdgcn_wmma_f32_16x16x32_bf16(false, a, false, b,
                                                 (short)0, c, false, false);
}

// Epilogue for one 16x16 C tile whose columns are (b, s) of this lane.
template<int HW, int WIMG, int FLAGS>
__device__ inline void epilogue_store(v8f acc, int mtile, int b, int s,
    const float* __restrict__ gamma, const float* __restrict__ beta,
    const float* __restrict__ resid, const float* __restrict__ maskp,
    float* __restrict__ Y, int Cout, int lane)
{
  const int mbase = mtile + ((lane >> 4) << 3);
  float mval = 1.0f;
  if (FLAGS & EPI_MASK) {                 // spatial gate: G=4, 7x7 cells of 8x8 px
    int h = s / WIMG, w = s % WIMG;
    int cg = mbase >> 7;                  // Cout/G = 128
    mval = maskp[(((size_t)b * 4 + cg) * 7 + (h >> 3)) * 7 + (w >> 3)];
  }
#pragma unroll
  for (int r = 0; r < 8; ++r) {
    int m = mbase + r;
    float v = acc[r] * gamma[m] + beta[m];
    if (FLAGS & EPI_MASK)  v *= mval;
    size_t idx = ((size_t)b * Cout + m) * HW + s;
    if (FLAGS & EPI_RESID) v += resid[idx];
    if (FLAGS & EPI_RELU)  v = fmaxf(v, 0.0f);
    Y[idx] = v;
  }
}

// ---------------------------------------------------------------------------
// 1x1 conv implicit GEMM.  Block = 8 waves = 128(M) x 64(N) tile.
// Each wave: 32x32 register tile = 2 A-frags x 2 B-frags -> 4 wmmas / K-step.
// M = Cout, N = B*HW, K = CIN (bf16, step 32).
// Requires: Cout % 128 == 0 or grid.y*128 == Cout; N % 64 == 0; CIN % 32 == 0.
// ---------------------------------------------------------------------------
template<int CIN, int HW, int WIMG, int FLAGS>
__global__ __launch_bounds__(256) void conv1x1_bn_wmma(
    const float* __restrict__ X, const float* __restrict__ W,
    const float* __restrict__ gamma, const float* __restrict__ beta,
    const float* __restrict__ resid, const float* __restrict__ maskp,
    float* __restrict__ Y, int Cout)
{
  const int lane = threadIdx.x & 31;
  const int wave = threadIdx.x >> 5;
  const int m0 = blockIdx.y * 128 + (wave >> 1) * 32;
  const int n0 = blockIdx.x * 64  + (wave & 1) * 32;

  const int l15   = lane & 15;
  const int akoff = (lane >> 4) << 4;   // A frag: 16 consecutive K at +0 / +16
  const int bkoff = (lane >> 4) << 4;   // B frag: 16 consecutive K at +0 / +16

  // two output-column groups (N subtiles) for this wave
  const int nA = n0 + l15,      nB = n0 + 16 + l15;
  const int bA = nA / HW,       sA = nA % HW;         // HW is constexpr -> cheap
  const int bB = nB / HW,       sB = nB % HW;
  // two output-row groups (M subtiles)
  const int rowA = m0 + l15,    rowB = m0 + 16 + l15;

  v8f acc00 = {}, acc01 = {}, acc10 = {}, acc11 = {};

  for (int k0 = 0; k0 < CIN; k0 += 32) {
    const float* wpA = W + (size_t)rowA * CIN + (k0 + akoff);
    const float* wpB = W + (size_t)rowB * CIN + (k0 + akoff);
    const float* xpA = X + ((size_t)bA * CIN + (k0 + bkoff)) * HW + sA;
    const float* xpB = X + ((size_t)bB * CIN + (k0 + bkoff)) * HW + sB;
    if (k0 + 32 < CIN) {
      __builtin_prefetch(xpA + (size_t)32 * HW, 0, 1);   // global_prefetch_b8
      __builtin_prefetch(xpB + (size_t)32 * HW, 0, 1);
    }
    v16bf a0, a1, b0, b1;
#pragma unroll
    for (int j = 0; j < 8; ++j) {
      a0[j] = (__bf16)wpA[j];  a0[8 + j] = (__bf16)wpA[8 + j];
      a1[j] = (__bf16)wpB[j];  a1[8 + j] = (__bf16)wpB[8 + j];
    }
#pragma unroll
    for (int j = 0; j < 16; ++j) {
      b0[j] = (__bf16)xpA[(size_t)j * HW];
      b1[j] = (__bf16)xpB[(size_t)j * HW];
    }
    acc00 = wmma_bf16(a0, b0, acc00);
    acc01 = wmma_bf16(a0, b1, acc01);
    acc10 = wmma_bf16(a1, b0, acc10);
    acc11 = wmma_bf16(a1, b1, acc11);
  }

  epilogue_store<HW, WIMG, FLAGS>(acc00, m0,      bA, sA, gamma, beta, resid, maskp, Y, Cout, lane);
  epilogue_store<HW, WIMG, FLAGS>(acc01, m0,      bB, sB, gamma, beta, resid, maskp, Y, Cout, lane);
  epilogue_store<HW, WIMG, FLAGS>(acc10, m0 + 16, bA, sA, gamma, beta, resid, maskp, Y, Cout, lane);
  epilogue_store<HW, WIMG, FLAGS>(acc11, m0 + 16, bB, sB, gamma, beta, resid, maskp, Y, Cout, lane);
}

// ---------------------------------------------------------------------------
// 3x3 conv (pad=1) implicit GEMM + bn + relu.  CIN = COUT = 128.
// Same 32x32 per-wave tiling; K loop = 9 taps x (128/32).
// ---------------------------------------------------------------------------
template<int STRIDE, int HIN, int WIN, int HOUT, int WOUT>
__global__ __launch_bounds__(256) void conv3x3_bn_wmma(
    const float* __restrict__ X, const float* __restrict__ W,
    const float* __restrict__ gamma, const float* __restrict__ beta,
    float* __restrict__ Y)
{
  constexpr int CIN = 128, COUT = 128;
  constexpr int HWo = HOUT * WOUT;
  constexpr int HWi = HIN * WIN;

  const int lane = threadIdx.x & 31;
  const int wave = threadIdx.x >> 5;
  const int m0 = (wave >> 1) * 32;                 // grid.y == 1 (COUT = 128)
  const int n0 = blockIdx.x * 64 + (wave & 1) * 32;

  const int l15   = lane & 15;
  const int akoff = (lane >> 4) << 4;
  const int bkoff = (lane >> 4) << 4;

  const int nA = n0 + l15,  nB = n0 + 16 + l15;
  const int bA = nA / HWo,  sA = nA % HWo;
  const int bB = nB / HWo,  sB = nB % HWo;
  const int ohA = sA / WOUT, owA = sA % WOUT;
  const int ohB = sB / WOUT, owB = sB % WOUT;
  const int rowA = m0 + l15, rowB = m0 + 16 + l15;

  v8f acc00 = {}, acc01 = {}, acc10 = {}, acc11 = {};

#pragma unroll
  for (int kh = 0; kh < 3; ++kh) {
#pragma unroll
    for (int kw = 0; kw < 3; ++kw) {
      const int tap = kh * 3 + kw;
      const int ihA = ohA * STRIDE - 1 + kh, iwA = owA * STRIDE - 1 + kw;
      const int ihB = ohB * STRIDE - 1 + kh, iwB = owB * STRIDE - 1 + kw;
      const bool vA = (ihA >= 0) && (ihA < HIN) && (iwA >= 0) && (iwA < WIN);
      const bool vB = (ihB >= 0) && (ihB < HIN) && (iwB >= 0) && (iwB < WIN);
      for (int k0 = 0; k0 < CIN; k0 += 32) {
        const float* wpA = W + ((size_t)rowA * CIN + (k0 + akoff)) * 9 + tap;
        const float* wpB = W + ((size_t)rowB * CIN + (k0 + akoff)) * 9 + tap;
        v16bf a0, a1, b0, b1;
#pragma unroll
        for (int j = 0; j < 8; ++j) {
          a0[j] = (__bf16)wpA[(size_t)j * 9];  a0[8 + j] = (__bf16)wpA[(size_t)(8 + j) * 9];
          a1[j] = (__bf16)wpB[(size_t)j * 9];  a1[8 + j] = (__bf16)wpB[(size_t)(8 + j) * 9];
        }
        if (vA) {
          const float* xp = X + (((size_t)bA * CIN + (k0 + bkoff)) * HIN + ihA) * WIN + iwA;
#pragma unroll
          for (int j = 0; j < 16; ++j) b0[j] = (__bf16)xp[(size_t)j * HWi];
        } else {
#pragma unroll
          for (int j = 0; j < 16; ++j) b0[j] = (__bf16)0.0f;
        }
        if (vB) {
          const float* xp = X + (((size_t)bB * CIN + (k0 + bkoff)) * HIN + ihB) * WIN + iwB;
#pragma unroll
          for (int j = 0; j < 16; ++j) b1[j] = (__bf16)xp[(size_t)j * HWi];
        } else {
#pragma unroll
          for (int j = 0; j < 16; ++j) b1[j] = (__bf16)0.0f;
        }
        acc00 = wmma_bf16(a0, b0, acc00);
        acc01 = wmma_bf16(a0, b1, acc01);
        acc10 = wmma_bf16(a1, b0, acc10);
        acc11 = wmma_bf16(a1, b1, acc11);
      }
    }
  }

  epilogue_store<HWo, WOUT, EPI_RELU>(acc00, m0,      bA, sA, gamma, beta, nullptr, nullptr, Y, COUT, lane);
  epilogue_store<HWo, WOUT, EPI_RELU>(acc01, m0,      bB, sB, gamma, beta, nullptr, nullptr, Y, COUT, lane);
  epilogue_store<HWo, WOUT, EPI_RELU>(acc10, m0 + 16, bA, sA, gamma, beta, nullptr, nullptr, Y, COUT, lane);
  epilogue_store<HWo, WOUT, EPI_RELU>(acc11, m0 + 16, bB, sB, gamma, beta, nullptr, nullptr, Y, COUT, lane);
}

// ---------------------------------------------------------------------------
// avgpool 3x3 stride 2 pad 1, constant /9 (matches reference)
// ---------------------------------------------------------------------------
template<int C, int HIN, int WIN, int HOUT, int WOUT>
__global__ void avgpool3s2_kernel(const float* __restrict__ X, float* __restrict__ Y)
{
  constexpr int total = 32 * C * HOUT * WOUT;
  int i = blockIdx.x * blockDim.x + threadIdx.x;
  if (i >= total) return;
  int wo = i % WOUT; int t = i / WOUT;
  int ho = t % HOUT; t /= HOUT;
  int c  = t % C;    int b = t / C;
  const float* p = X + ((size_t)b * C + c) * HIN * WIN;
  float sum = 0.0f;
#pragma unroll
  for (int dh = 0; dh < 3; ++dh) {
    int ih = ho * 2 - 1 + dh;
    if (ih < 0 || ih >= HIN) continue;
#pragma unroll
    for (int dw = 0; dw < 3; ++dw) {
      int iw = wo * 2 - 1 + dw;
      if (iw < 0 || iw >= WIN) continue;
      sum += p[(size_t)ih * WIN + iw];
    }
  }
  Y[i] = sum * (1.0f / 9.0f);
}

// ---------------------------------------------------------------------------
// fused = relu( bilinear_up2x(x_base) + x_refine ), half-pixel centers
// ---------------------------------------------------------------------------
__global__ void fuse_upsample_kernel(const float* __restrict__ xb,
                                     const float* __restrict__ rf,
                                     float* __restrict__ fused)
{
  const int total = 32 * 512 * 56 * 56;
  int i = blockIdx.x * blockDim.x + threadIdx.x;
  if (i >= total) return;
  int w = i % 56; int t = i / 56;
  int h = t % 56; t /= 56;
  int c = t % 512; int b = t / 512;
  float sy = 0.5f * h - 0.25f;
  float sx = 0.5f * w - 0.25f;
  int y0 = (int)floorf(sy); float wy = sy - (float)y0;
  int x0 = (int)floorf(sx); float wx = sx - (float)x0;
  int y0c = y0 < 0 ? 0 : (y0 > 27 ? 27 : y0);
  int y1 = y0 + 1; int y1c = y1 < 0 ? 0 : (y1 > 27 ? 27 : y1);
  int x0c = x0 < 0 ? 0 : (x0 > 27 ? 27 : x0);
  int x1 = x0 + 1; int x1c = x1 < 0 ? 0 : (x1 > 27 ? 27 : x1);
  const float* p = xb + ((size_t)b * 512 + c) * 784;
  float v00 = p[y0c * 28 + x0c], v01 = p[y0c * 28 + x1c];
  float v10 = p[y1c * 28 + x0c], v11 = p[y1c * 28 + x1c];
  float v = (1.0f - wy) * ((1.0f - wx) * v00 + wx * v01)
          +          wy * ((1.0f - wx) * v10 + wx * v11);
  fused[i] = fmaxf(v + rf[i], 0.0f);
}

// ---------------------------------------------------------------------------
// Mask branch: grouped 3x3 conv (512 -> 16, G=4) + bn + relu   (small, scalar)
// ---------------------------------------------------------------------------
__global__ void mask_gconv_kernel(const float* __restrict__ xb, const float* __restrict__ W,
                                  const float* __restrict__ g, const float* __restrict__ bt,
                                  float* __restrict__ gmap)
{
  const int total = 32 * 16 * 28 * 28;
  int i = blockIdx.x * blockDim.x + threadIdx.x;
  if (i >= total) return;
  int w = i % 28; int t = i / 28;
  int h = t % 28; t /= 28;
  int oc = t % 16; int b = t / 16;
  int cin0 = (oc >> 2) * 128;
  float sum = 0.0f;
  for (int kh = 0; kh < 3; ++kh) {
    int ih = h - 1 + kh;
    if (ih < 0 || ih >= 28) continue;
    for (int kw = 0; kw < 3; ++kw) {
      int iw = w - 1 + kw;
      if (iw < 0 || iw >= 28) continue;
      const float* xp = xb + (((size_t)b * 512 + cin0) * 28 + ih) * 28 + iw;
      const float* wp = W + (size_t)oc * 128 * 9 + kh * 3 + kw;
      for (int ci = 0; ci < 128; ++ci)
        sum += xp[(size_t)ci * 784] * wp[(size_t)ci * 9];
    }
  }
  gmap[i] = fmaxf(sum * g[oc] + bt[oc], 0.0f);
}

// (32,16,28,28) -> mean over 4x4 blocks -> (32,16,7,7)
__global__ void mask_pool_kernel(const float* __restrict__ gmap, float* __restrict__ gp)
{
  const int total = 32 * 16 * 49;
  int i = blockIdx.x * blockDim.x + threadIdx.x;
  if (i >= total) return;
  int j7 = i % 7; int t = i / 7;
  int i7 = t % 7; t /= 7;
  int c  = t % 16; int b = t / 16;
  const float* p = gmap + ((size_t)b * 16 + c) * 784;
  float s = 0.0f;
#pragma unroll
  for (int di = 0; di < 4; ++di)
#pragma unroll
    for (int dj = 0; dj < 4; ++dj)
      s += p[(i7 * 4 + di) * 28 + (j7 * 4 + dj)];
  gp[i] = s * (1.0f / 16.0f);
}

// grouped 1x1 FC (16->8, G=4) + bias + gumbel -> hard mask (argmax over axis 1)
__global__ void mask_gate_kernel(const float* __restrict__ gp, const float* __restrict__ fcw,
                                 const float* __restrict__ fcb, const float* __restrict__ noise,
                                 float* __restrict__ maskb, float* __restrict__ outmask)
{
  const int total = 32 * 4 * 49;
  int i = blockIdx.x * blockDim.x + threadIdx.x;
  if (i >= total) return;
  int j7 = i % 7; int t = i / 7;
  int i7 = t % 7; t /= 7;
  int g  = t % 4; int b = t / 4;
  int oc0 = g, oc1 = 4 + g;                    // (c2, g) -> channel c2*4+g
  float l0 = fcb[oc0], l1 = fcb[oc1];
#pragma unroll
  for (int ci = 0; ci < 4; ++ci) {
    l0 += fcw[oc0 * 4 + ci] * gp[(((size_t)b * 16 + (oc0 >> 1) * 4 + ci) * 7 + i7) * 7 + j7];
    l1 += fcw[oc1 * 4 + ci] * gp[(((size_t)b * 16 + (oc1 >> 1) * 4 + ci) * 7 + i7) * 7 + j7];
  }
  float a0 = l0 + noise[((((size_t)b * 2 + 0) * 4 + g) * 7 + i7) * 7 + j7];
  float a1 = l1 + noise[((((size_t)b * 2 + 1) * 4 + g) * 7 + i7) * 7 + j7];
  float mv = (a1 > a0) ? 1.0f : 0.0f;
  maskb[i]   = mv;
  outmask[i] = mv;
}

// ---------------------------------------------------------------------------
// Orchestration
// ---------------------------------------------------------------------------
extern "C" void kernel_launch(void* const* d_in, const int* in_sizes, int n_in,
                              void* d_out, int out_size, void* d_ws, size_t ws_size,
                              hipStream_t stream)
{
  (void)in_sizes; (void)n_in; (void)out_size; (void)ws_size;
  const float* x     = (const float*)d_in[0];
  const float* gnoise= (const float*)d_in[1];
  const float* b_w1  = (const float*)d_in[2];
  const float* b_g1  = (const float*)d_in[3];
  const float* b_b1  = (const float*)d_in[4];
  const float* b_w2  = (const float*)d_in[5];
  const float* b_g2  = (const float*)d_in[6];
  const float* b_b2  = (const float*)d_in[7];
  const float* b_w3  = (const float*)d_in[8];
  const float* b_g3  = (const float*)d_in[9];
  const float* b_b3  = (const float*)d_in[10];
  const float* b_dw  = (const float*)d_in[11];
  const float* b_dg  = (const float*)d_in[12];
  const float* b_db  = (const float*)d_in[13];
  const float* r_w1  = (const float*)d_in[14];
  const float* r_g1  = (const float*)d_in[15];
  const float* r_b1  = (const float*)d_in[16];
  const float* r_w2  = (const float*)d_in[17];
  const float* r_g2  = (const float*)d_in[18];
  const float* r_b2  = (const float*)d_in[19];
  const float* r_w3  = (const float*)d_in[20];
  const float* r_g3  = (const float*)d_in[21];
  const float* r_b3  = (const float*)d_in[22];
  const float* r_dw  = (const float*)d_in[23];
  const float* r_dg  = (const float*)d_in[24];
  const float* r_db  = (const float*)d_in[25];
  const float* m_w   = (const float*)d_in[26];
  const float* m_g   = (const float*)d_in[27];
  const float* m_b   = (const float*)d_in[28];
  const float* m_fcw = (const float*)d_in[29];
  const float* m_fcb = (const float*)d_in[30];
  const float* f_w1  = (const float*)d_in[31];
  const float* f_g1  = (const float*)d_in[32];
  const float* f_b1  = (const float*)d_in[33];
  const float* f_w2  = (const float*)d_in[34];
  const float* f_g2  = (const float*)d_in[35];
  const float* f_b2  = (const float*)d_in[36];
  const float* f_w3  = (const float*)d_in[37];
  const float* f_g3  = (const float*)d_in[38];
  const float* f_b3  = (const float*)d_in[39];

  const int B = 32, CIN = 256, COUT = 512, PL = 128;
  const int HW56 = 3136, HW28 = 784;
  const int N56 = B * HW56;        // 100352
  const int N28 = B * HW28;        // 25088

  // workspace layout (floats)
  float* ws     = (float*)d_ws;
  float* pooled = ws;                         // up to 12,845,056
  float* buf1   = pooled + 12845056;          // 12,845,056
  float* buf2   = buf1   + 12845056;          //  3,211,264
  float* dwb    = buf2   + 3211264;           // 12,845,056
  float* xbase  = dwb    + 12845056;          // 12,845,056
  float* refine = xbase  + 12845056;          // 51,380,224
  float* fused  = refine + 51380224;          // 51,380,224
  float* gmap   = fused  + 51380224;          //    401,408
  float* gpool  = gmap   + 401408;            //     25,088
  float* maskb  = gpool  + 25088;             //      6,272

  float* outT  = (float*)d_out;               // 32*512*28*28
  float* outM  = outT + 12845056;             // 32*4*7*7

  dim3 blk(256);

  // 1) pooled_x = avgpool3s2(x)           (32,256,28,28)
  { int tot = B * CIN * HW28;
    avgpool3s2_kernel<256, 56, 56, 28, 28><<<dim3((tot + 255) / 256), blk, 0, stream>>>(x, pooled); }
  // 2) b1 = relu(bn(conv1x1(x, b_w1)))    (32,128,56,56)
  conv1x1_bn_wmma<256, 3136, 56, EPI_RELU><<<dim3(N56 / 64, 1), blk, 0, stream>>>(
      x, b_w1, b_g1, b_b1, nullptr, nullptr, buf1, PL);
  // 3) b2 = relu(bn(conv3x3 s2(b1)))      (32,128,28,28)
  conv3x3_bn_wmma<2, 56, 56, 28, 28><<<dim3(N28 / 64, 1), blk, 0, stream>>>(
      buf1, b_w2, b_g2, b_b2, buf2);
  // 4) dwb = bn(conv1x1(pooled_x, b_dw))  (32,512,28,28)
  conv1x1_bn_wmma<256, 784, 28, 0><<<dim3(N28 / 64, COUT / 128), blk, 0, stream>>>(
      pooled, b_dw, b_dg, b_db, nullptr, nullptr, dwb, COUT);
  // 5) x_base = bn(conv1x1(b2, b_w3)) + dwb
  conv1x1_bn_wmma<128, 784, 28, EPI_RESID><<<dim3(N28 / 64, COUT / 128), blk, 0, stream>>>(
      buf2, b_w3, b_g3, b_b3, dwb, nullptr, xbase, COUT);
  // 6-8) mask branch
  { int tot = 32 * 16 * 28 * 28;
    mask_gconv_kernel<<<dim3((tot + 255) / 256), blk, 0, stream>>>(xbase, m_w, m_g, m_b, gmap); }
  { int tot = 32 * 16 * 49;
    mask_pool_kernel<<<dim3((tot + 255) / 256), blk, 0, stream>>>(gmap, gpool); }
  { int tot = 32 * 4 * 49;
    mask_gate_kernel<<<dim3((tot + 255) / 256), blk, 0, stream>>>(gpool, m_fcw, m_fcb, gnoise, maskb, outM); }
  // 9) r1 = relu(bn(conv1x1(x, r_w1)))    (32,128,56,56)
  conv1x1_bn_wmma<256, 3136, 56, EPI_RELU><<<dim3(N56 / 64, 1), blk, 0, stream>>>(
      x, r_w1, r_g1, r_b1, nullptr, nullptr, buf1, PL);
  // 10) r2 = relu(bn(conv3x3 s1(r1)))     (32,128,56,56)  -> reuse `pooled`
  conv3x3_bn_wmma<1, 56, 56, 56, 56><<<dim3(N56 / 64, 1), blk, 0, stream>>>(
      buf1, r_w2, r_g2, r_b2, pooled);
  // 11) refine = bn(conv1x1(x, r_dw))     (32,512,56,56)
  conv1x1_bn_wmma<256, 3136, 56, 0><<<dim3(N56 / 64, COUT / 128), blk, 0, stream>>>(
      x, r_dw, r_dg, r_db, nullptr, nullptr, refine, COUT);
  // 12) refine = bn(conv1x1(r2, r_w3)) * mask_full + refine  (in-place resid)
  conv1x1_bn_wmma<128, 3136, 56, EPI_MASK | EPI_RESID><<<dim3(N56 / 64, COUT / 128), blk, 0, stream>>>(
      pooled, r_w3, r_g3, r_b3, refine, maskb, refine, COUT);
  // 13) fused = relu(upsample2x(x_base) + refine)
  { int tot = B * COUT * HW56;
    fuse_upsample_kernel<<<dim3((tot + 255) / 256), blk, 0, stream>>>(xbase, refine, fused); }
  // 14) pooled_fused = avgpool3s2(fused)  (32,512,28,28)
  { int tot = B * COUT * HW28;
    avgpool3s2_kernel<512, 56, 56, 28, 28><<<dim3((tot + 255) / 256), blk, 0, stream>>>(fused, pooled); }
  // 15) f1 = relu(bn(conv1x1(fused, f_w1)))
  conv1x1_bn_wmma<512, 3136, 56, EPI_RELU><<<dim3(N56 / 64, 1), blk, 0, stream>>>(
      fused, f_w1, f_g1, f_b1, nullptr, nullptr, buf1, PL);
  // 16) f2 = relu(bn(conv3x3 s2(f1)))
  conv3x3_bn_wmma<2, 56, 56, 28, 28><<<dim3(N28 / 64, 1), blk, 0, stream>>>(
      buf1, f_w2, f_g2, f_b2, buf2);
  // 17) out = relu(bn(conv1x1(f2, f_w3)) + pooled_fused)
  conv1x1_bn_wmma<128, 784, 28, EPI_RESID | EPI_RELU><<<dim3(N28 / 64, COUT / 128), blk, 0, stream>>>(
      buf2, f_w3, f_g3, f_b3, pooled, nullptr, outT, COUT);
}
